// AttentionBlock_6846177870078
// MI455X (gfx1250) — matrix operand order repaired and checked by reading, not compiled, not run on hardware
//
#include <hip/hip_runtime.h>
#include <hip/hip_bf16.h>

// ---------------------------------------------------------------------------
// CDNA5 (gfx1250): bf16 WMMA GEMMs + flash attention, double-buffered LDS
// staging with gfx1250 async global->LDS copies when the toolchain has them.
// ---------------------------------------------------------------------------

#if defined(__has_builtin)
#  if __has_builtin(__builtin_amdgcn_global_load_async_to_lds_b128) && \
      __has_builtin(__builtin_amdgcn_s_wait_asynccnt)
#    define USE_ASYNC_LDS 1
#  endif
#endif
#ifndef USE_ASYNC_LDS
#  define USE_ASYNC_LDS 0
#endif

typedef __attribute__((ext_vector_type(16))) __bf16 bf16x16;
typedef __attribute__((ext_vector_type(8)))  float  f32x8;
typedef int v4i __attribute__((vector_size(16)));            // matches builtin param
typedef __attribute__((address_space(1))) v4i* gas_b128;     // global int4*
typedef __attribute__((address_space(3))) v4i* las_b128;     // LDS int4*

struct Frag { unsigned int u[8]; };   // 16 bf16 packed as 8 dwords

__device__ __forceinline__ f32x8 zero_v8() {
    f32x8 z;
    #pragma unroll
    for (int i = 0; i < 8; ++i) z[i] = 0.0f;
    return z;
}

__device__ __forceinline__ f32x8 wmma_bf16(const Frag& a, const Frag& b, f32x8 c) {
    return __builtin_amdgcn_wmma_f32_16x16x32_bf16(
        /*neg_a=*/false, __builtin_bit_cast(bf16x16, a),
        /*neg_b=*/false, __builtin_bit_cast(bf16x16, b),
        /*c_mod=*/(short)0, c, /*reuse_a=*/false, /*reuse_b=*/false);
}

// round-to-nearest-even f32 -> bf16 (raw u16)
__device__ __forceinline__ unsigned short f2bf(float f) {
    unsigned int u = __builtin_bit_cast(unsigned int, f);
    u = (u + 0x7FFFu + ((u >> 16) & 1u)) >> 16;
    return (unsigned short)u;
}

// A-fragment k-offset within a 32-wide K tile (16-bit A layout, ISA 7.12.2)
__device__ __forceinline__ int kA(int v, int half) {
    return (v & 3) * 2 + (v >> 2) * 16 + 8 * half;
}
// B-fragment k-offset within a 32-wide K tile (16-bit B layout)
__device__ __forceinline__ int kB(int v, int half) {
    return (v & 3) * 2 + (v >> 2) * 16 + 8 * half;
}

// async global->LDS copy (gfx1250 GLOBAL_LOAD_ASYNC_TO_LDS_B128) with fallback
__device__ __forceinline__ void async_b128(const unsigned short* g, unsigned short* l) {
#if USE_ASYNC_LDS
    __builtin_amdgcn_global_load_async_to_lds_b128((gas_b128)g, (las_b128)l, 0, 0);
#else
    (void)g; (void)l;
#endif
}
__device__ __forceinline__ void async_wait() {
#if USE_ASYNC_LDS
    __builtin_amdgcn_s_wait_asynccnt(0);
#endif
}

// reductions across the 16 lanes of a half-wave (C-layout row lives in one
// VGPR across 16 lanes)
__device__ __forceinline__ float rowmax16(float v) {
    #pragma unroll
    for (int m = 1; m <= 8; m <<= 1) v = fmaxf(v, __shfl_xor(v, m, 32));
    return v;
}
__device__ __forceinline__ float rowsum16(float v) {
    #pragma unroll
    for (int m = 1; m <= 8; m <<= 1) v += __shfl_xor(v, m, 32);
    return v;
}

// ---------------------------------------------------------------------------
// f32 -> bf16 elementwise convert (weights)
// ---------------------------------------------------------------------------
__global__ void cvt_f32_bf16(const float* __restrict__ in,
                             unsigned short* __restrict__ out, int n) {
    int i = blockIdx.x * blockDim.x + threadIdx.x;
    if (i < n) out[i] = f2bf(in[i]);
}

// ---------------------------------------------------------------------------
// T5 relative-position bias table: tab[t][h], rel = t - (m-1).
// ---------------------------------------------------------------------------
__global__ void relbias_kernel(const float* __restrict__ emb,   // [32,16]
                               float* __restrict__ tab, int n, int m) {
    int idx = blockIdx.x * blockDim.x + threadIdx.x;
    int total = (n + m - 1) * 16;
    if (idx >= total) return;
    int t = idx >> 4, h = idx & 15;
    int rel = t - (m - 1);
    int ret = (rel >= 0) ? 16 : 0;              // NUM_BUCKETS/2
    int na = rel < 0 ? -rel : rel;
    int bucket;
    if (na < 8) {                               // max_exact = 8
        bucket = ret + na;
    } else {
        float f = __logf((float)na * 0.125f) * (8.0f / __logf(16.0f));
        int vl = 8 + (int)f;
        bucket = ret + (vl < 15 ? vl : 15);
    }
    tab[idx] = emb[bucket * 16 + h];
}

// ---------------------------------------------------------------------------
// LayerNorm over F=1024, f32 in -> bf16 out. One block per row.
// ---------------------------------------------------------------------------
__global__ __launch_bounds__(256)
void ln_bf16_kernel(const float* __restrict__ x, const float* __restrict__ g,
                    const float* __restrict__ bta, unsigned short* __restrict__ out) {
    __shared__ float red[256];
    const int row = blockIdx.x, tid = threadIdx.x;
    const float* xp = x + (size_t)row * 1024;

    float s = 0.f;
    for (int j = tid; j < 1024; j += 256) s += xp[j];
    red[tid] = s; __syncthreads();
    for (int st = 128; st > 0; st >>= 1) {
        if (tid < st) red[tid] += red[tid + st];
        __syncthreads();
    }
    float mean = red[0] * (1.0f / 1024.0f);
    __syncthreads();

    float v = 0.f;
    for (int j = tid; j < 1024; j += 256) { float d = xp[j] - mean; v += d * d; }
    red[tid] = v; __syncthreads();
    for (int st = 128; st > 0; st >>= 1) {
        if (tid < st) red[tid] += red[tid + st];
        __syncthreads();
    }
    float rstd = rsqrtf(red[0] * (1.0f / 1024.0f) + 1e-5f);

    unsigned short* op = out + (size_t)row * 1024;
    for (int j = tid; j < 1024; j += 256)
        op[j] = f2bf((xp[j] - mean) * rstd * g[j] + bta[j]);
}

// ---------------------------------------------------------------------------
// bf16 GEMM: C[M,N] = A[M,K] @ B[K,N]. 128x128 block, BK=32, double-buffered
// LDS staging (A via async copy, B register-transposed). 8 waves x (2x4)
// WMMA tiles. mode 0: bf16 store. mode 1: f32 + bias[col] + residual.
// ---------------------------------------------------------------------------
#define BM 128
#define BN 128
#define BK 32

__global__ __launch_bounds__(256)
void gemm_bf16_kernel(const unsigned short* __restrict__ A,
                      const unsigned short* __restrict__ B,
                      int M, int N, int K,
                      unsigned short* __restrict__ Cb,
                      float* __restrict__ Cf,
                      const float* __restrict__ bias,
                      const float* __restrict__ res,
                      int mode) {
    __shared__ unsigned short lA[2][BM][BK];     // 16 KB
    __shared__ unsigned short lBT[2][BN][BK];    // 16 KB (B transposed)

    const int tid = threadIdx.x;
    const int wid = tid >> 5, lane = tid & 31;
    const int half = lane >> 4, l16 = lane & 15;
    const int wm = wid >> 1, wn = wid & 1;       // 4x2 wave grid
    const int m0 = blockIdx.y * BM, n0 = blockIdx.x * BN;

    // staging geometry (per thread, fixed across K-steps)
    const int ar = tid >> 2, ac8 = (tid & 3) * 8;    // A rows ar, ar+64
    const int bk = tid >> 4, bc8 = (tid & 15) * 8;   // B k-rows bk, bk+16
    const unsigned short* aG0 = A + (size_t)(m0 + ar) * K + ac8;
    const unsigned short* aG1 = A + (size_t)(m0 + ar + 64) * K + ac8;
    const unsigned short* bG0 = B + (size_t)bk * N + n0 + bc8;
    const unsigned short* bG1 = B + (size_t)(bk + 16) * N + n0 + bc8;

    struct Stage {
        uint4 b0, b1;
#if !USE_ASYNC_LDS
        uint4 a0, a1;
#endif
    };
    auto issue = [&](int kt, int buf, Stage& st) {
        const size_t ka = (size_t)kt * BK;
        const size_t kb_ = (size_t)kt * BK * N;
#if USE_ASYNC_LDS
        async_b128(aG0 + ka, &lA[buf][ar][ac8]);
        async_b128(aG1 + ka, &lA[buf][ar + 64][ac8]);
#else
        st.a0 = *(const uint4*)(aG0 + ka);
        st.a1 = *(const uint4*)(aG1 + ka);
#endif
        st.b0 = *(const uint4*)(bG0 + kb_);
        st.b1 = *(const uint4*)(bG1 + kb_);
    };
    auto commit = [&](int buf, Stage& st) {
#if !USE_ASYNC_LDS
        *(uint4*)&lA[buf][ar][ac8]      = st.a0;
        *(uint4*)&lA[buf][ar + 64][ac8] = st.a1;
#endif
        const unsigned short* e0 = (const unsigned short*)&st.b0;
        const unsigned short* e1 = (const unsigned short*)&st.b1;
        #pragma unroll
        for (int j = 0; j < 8; ++j) lBT[buf][bc8 + j][bk] = e0[j];
        #pragma unroll
        for (int j = 0; j < 8; ++j) lBT[buf][bc8 + j][bk + 16] = e1[j];
        async_wait();
    };

    f32x8 acc[2][4];
    #pragma unroll
    for (int i = 0; i < 2; ++i)
        #pragma unroll
        for (int j = 0; j < 4; ++j) acc[i][j] = zero_v8();

    const int nk = K / BK;
    {
        Stage st;
        issue(0, 0, st);
        commit(0, st);
    }
    __syncthreads();

    int cur = 0;
    for (int kt = 0; kt < nk; ++kt) {
        Stage st;
        const bool nxt = (kt + 1 < nk);
        if (nxt) issue(kt + 1, cur ^ 1, st);     // loads fly under the WMMAs

        Frag af[2], bfr[4];
        #pragma unroll
        for (int mt = 0; mt < 2; ++mt) {
            const unsigned short* rp = &lA[cur][wm * 32 + mt * 16 + l16][0];
            #pragma unroll
            for (int v = 0; v < 8; ++v)
                af[mt].u[v] = *(const unsigned int*)(rp + kA(v, half));
        }
        #pragma unroll
        for (int nt = 0; nt < 4; ++nt) {
            const unsigned short* rp = &lBT[cur][wn * 64 + nt * 16 + l16][0];
            #pragma unroll
            for (int v = 0; v < 8; ++v)
                bfr[nt].u[v] = *(const unsigned int*)(rp + kB(v, half));
        }
        #pragma unroll
        for (int mt = 0; mt < 2; ++mt)
            #pragma unroll
            for (int nt = 0; nt < 4; ++nt)
                acc[mt][nt] = wmma_bf16(af[mt], bfr[nt], acc[mt][nt]);

        if (nxt) {
            commit(cur ^ 1, st);
            __syncthreads();
        }
        cur ^= 1;
    }

    #pragma unroll
    for (int mt = 0; mt < 2; ++mt)
        #pragma unroll
        for (int nt = 0; nt < 4; ++nt)
            #pragma unroll
            for (int r = 0; r < 8; ++r) {
                int row = m0 + wm * 32 + mt * 16 + r + 8 * half;
                int col = n0 + wn * 64 + nt * 16 + l16;
                float val = acc[mt][nt][r];
                if (mode == 0)
                    Cb[(size_t)row * N + col] = f2bf(val);
                else
                    Cf[(size_t)row * N + col] =
                        val + bias[col] + res[(size_t)row * N + col];
            }
}

// ---------------------------------------------------------------------------
// Flash attention, one (b,h,64-q-rows) block, 4 waves x 16 q-rows.
// K and V^T blocks (64x64 bf16 each) staged cooperatively into LDS,
// double-buffered; P staged through LDS for C-layout -> A-layout.
//   q stride 1024 (MID), kv stride 2048 (2*MID), V at +1024.
// ---------------------------------------------------------------------------
__global__ __launch_bounds__(128)
void attn_kernel(const unsigned short* __restrict__ qm,
                 const unsigned short* __restrict__ kvm,
                 const float* __restrict__ tab,      // [(n+m-1), 16]
                 unsigned short* __restrict__ om,    // [b, n, 1024]
                 int n, int m) {
    __shared__ unsigned short lK[2][64][64];    // 16 KB, [j_local][d]
    __shared__ unsigned short lVT[2][64][64];   // 16 KB, [d][m_local]
    __shared__ unsigned short P[4][16][64];     // 8 KB, per-wave slice

    const int tid = threadIdx.x;
    const int wid = tid >> 5, lane = tid & 31;
    const int half = lane >> 4, l16 = lane & 15;
    const int q0 = blockIdx.x * 64 + wid * 16;
    const int h = blockIdx.y, b = blockIdx.z;

    const unsigned short* qb = qm + (size_t)b * n * 1024 + h * 64;
    const unsigned short* kb = kvm + (size_t)b * m * 2048 + h * 64;
    const unsigned short* vb = kb + 1024;
    unsigned short* ob = om + (size_t)b * n * 1024 + h * 64;

    // staging geometry: 128 threads x 4 iters cover 64 rows x 64 bf16
    const int sr  = tid >> 3;          // row base 0..15 (+16*it)
    const int sc8 = (tid & 7) * 8;     // 8-element column chunk

    struct VStage {
        uint4 v[4];
#if !USE_ASYNC_LDS
        uint4 k[4];
#endif
    };
    auto issue = [&](int j0, int buf, VStage& st) {
        #pragma unroll
        for (int it = 0; it < 4; ++it) {
            const int row = sr + 16 * it;
#if USE_ASYNC_LDS
            async_b128(kb + (size_t)(j0 + row) * 2048 + sc8, &lK[buf][row][sc8]);
#else
            st.k[it] = *(const uint4*)(kb + (size_t)(j0 + row) * 2048 + sc8);
#endif
            st.v[it] = *(const uint4*)(vb + (size_t)(j0 + row) * 2048 + sc8);
        }
    };
    auto commit = [&](int buf, VStage& st) {
        #pragma unroll
        for (int it = 0; it < 4; ++it) {
            const int row = sr + 16 * it;
#if !USE_ASYNC_LDS
            *(uint4*)&lK[buf][row][sc8] = st.k[it];
#endif
            const unsigned short* e = (const unsigned short*)&st.v[it];
            #pragma unroll
            for (int j = 0; j < 8; ++j) lVT[buf][sc8 + j][row] = e[j];
        }
        async_wait();
    };

    // Q fragments (A layout), d = 64 -> two 32-wide K steps
    Frag qf[2];
    {
        const unsigned short* rp = qb + (size_t)(q0 + l16) * 1024;
        #pragma unroll
        for (int ks = 0; ks < 2; ++ks)
            #pragma unroll
            for (int v = 0; v < 8; ++v)
                qf[ks].u[v] = *(const unsigned int*)(rp + 32 * ks + kA(v, half));
    }

    float mi[8], li[8];
    #pragma unroll
    for (int r = 0; r < 8; ++r) { mi[r] = -3.0e38f; li[r] = 0.f; }
    f32x8 O[4];
    #pragma unroll
    for (int t = 0; t < 4; ++t) O[t] = zero_v8();

    const float scale = 0.125f;   // 64^-0.5

    {
        VStage st;
        issue(0, 0, st);
        commit(0, st);
    }
    __syncthreads();

    int cur = 0;
    const int nblk = m / 64;
    for (int jb = 0; jb < nblk; ++jb) {
        const int j0 = jb * 64;
        VStage st;
        const bool nxt = (jb + 1 < nblk);
        if (nxt) issue(j0 + 64, cur ^ 1, st);

        // ---- S = Q K^T for a 16x64 stripe ----
        f32x8 S[4];
        #pragma unroll
        for (int nt = 0; nt < 4; ++nt) {
            S[nt] = zero_v8();
            const unsigned short* rp = &lK[cur][nt * 16 + l16][0];
            #pragma unroll
            for (int ks = 0; ks < 2; ++ks) {
                Frag bk2;   // B = K^T: pairs over d contiguous in lK rows
                #pragma unroll
                for (int v = 0; v < 8; ++v)
                    bk2.u[v] = *(const unsigned int*)(rp + 32 * ks + kB(v, half));
                S[nt] = wmma_bf16(qf[ks], bk2, S[nt]);
            }
        }
        // ---- bias + scale ----
        #pragma unroll
        for (int nt = 0; nt < 4; ++nt)
            #pragma unroll
            for (int r = 0; r < 8; ++r) {
                int i = q0 + r + 8 * half;
                int j = j0 + nt * 16 + l16;
                S[nt][r] = (S[nt][r] + tab[(size_t)(j - i + n - 1) * 16 + h]) * scale;
            }
        // ---- online softmax ----
        float alpha[8];
        #pragma unroll
        for (int r = 0; r < 8; ++r) {
            float mx = fmaxf(fmaxf(S[0][r], S[1][r]), fmaxf(S[2][r], S[3][r]));
            mx = rowmax16(mx);
            float mn = fmaxf(mi[r], mx);
            alpha[r] = __expf(mi[r] - mn);
            mi[r] = mn;
        }
        #pragma unroll
        for (int r = 0; r < 8; ++r) {
            float rs = 0.f;
            #pragma unroll
            for (int nt = 0; nt < 4; ++nt) {
                float p = __expf(S[nt][r] - mi[r]);
                rs += p;
                P[wid][r + 8 * half][nt * 16 + l16] = f2bf(p);
            }
            rs = rowsum16(rs);
            li[r] = li[r] * alpha[r] + rs;
            #pragma unroll
            for (int t = 0; t < 4; ++t) O[t][r] *= alpha[r];
        }
        // ---- P (A layout from LDS) @ V (B layout from lVT) ----
        Frag pf[2];
        const unsigned short* pr = &P[wid][l16][0];
        #pragma unroll
        for (int ks = 0; ks < 2; ++ks)
            #pragma unroll
            for (int v = 0; v < 8; ++v)
                pf[ks].u[v] = *(const unsigned int*)(pr + 32 * ks + kA(v, half));

        #pragma unroll
        for (int td = 0; td < 4; ++td) {
            const unsigned short* rp = &lVT[cur][td * 16 + l16][0];
            #pragma unroll
            for (int ks = 0; ks < 2; ++ks) {
                Frag vf;   // pairs over m contiguous in lVT rows
                #pragma unroll
                for (int v = 0; v < 8; ++v)
                    vf.u[v] = *(const unsigned int*)(rp + 32 * ks + kB(v, half));
                O[td] = wmma_bf16(pf[ks], vf, O[td]);
            }
        }

        if (nxt) {
            commit(cur ^ 1, st);
            __syncthreads();
        }
        cur ^= 1;
    }

    // ---- normalize and store bf16 head output ----
    #pragma unroll
    for (int td = 0; td < 4; ++td)
        #pragma unroll
        for (int r = 0; r < 8; ++r) {
            int i = q0 + r + 8 * half;
            int d = td * 16 + l16;
            ob[(size_t)i * 1024 + d] = f2bf(O[td][r] / li[r]);
        }
}

// ---------------------------------------------------------------------------
// Host-side orchestration
// ---------------------------------------------------------------------------
extern "C" void kernel_launch(void* const* d_in, const int* in_sizes, int n_in,
                              void* d_out, int out_size, void* d_ws, size_t ws_size,
                              hipStream_t stream) {
    (void)in_sizes; (void)n_in; (void)out_size; (void)ws_size;

    const int B = 2, N = 2048, M = 512, F = 1024, MID = 1024, H = 16;
    const int rowsX = B * N;   // 4096
    const int rowsC = B * M;   // 1024

    const float* x        = (const float*)d_in[0];
    const float* context  = (const float*)d_in[1];
    const float* sa_ng    = (const float*)d_in[2];
    const float* sa_nb    = (const float*)d_in[3];
    const float* sa_ncg   = (const float*)d_in[4];
    const float* sa_ncb   = (const float*)d_in[5];
    const float* sa_wq    = (const float*)d_in[6];
    const float* sa_wkv   = (const float*)d_in[7];
    const float* sa_wo    = (const float*)d_in[8];
    const float* sa_bo    = (const float*)d_in[9];
    const float* sa_rel   = (const float*)d_in[10];
    const float* ca_ng    = (const float*)d_in[11];
    const float* ca_nb    = (const float*)d_in[12];
    const float* ca_ncg   = (const float*)d_in[13];
    const float* ca_ncb   = (const float*)d_in[14];
    const float* ca_wq    = (const float*)d_in[15];
    const float* ca_wkv   = (const float*)d_in[16];
    const float* ca_wo    = (const float*)d_in[17];
    const float* ca_bo    = (const float*)d_in[18];
    const float* ca_rel   = (const float*)d_in[19];
    float* out = (float*)d_out;

    size_t off = 0;
    auto alloc = [&](size_t bytes) -> char* {
        off = (off + 255) & ~(size_t)255;
        char* p = (char*)d_ws + off;
        off += bytes;
        return p;
    };
    unsigned short* wq_s  = (unsigned short*)alloc((size_t)F * MID * 2);
    unsigned short* wkv_s = (unsigned short*)alloc((size_t)F * 2 * MID * 2);
    unsigned short* wo_s  = (unsigned short*)alloc((size_t)MID * F * 2);
    unsigned short* wq_c  = (unsigned short*)alloc((size_t)F * MID * 2);
    unsigned short* wkv_c = (unsigned short*)alloc((size_t)F * 2 * MID * 2);
    unsigned short* wo_c  = (unsigned short*)alloc((size_t)MID * F * 2);
    unsigned short* xn    = (unsigned short*)alloc((size_t)rowsX * F * 2);
    unsigned short* cn    = (unsigned short*)alloc((size_t)rowsX * F * 2);
    unsigned short* cnx   = (unsigned short*)alloc((size_t)rowsC * F * 2);
    unsigned short* qb    = (unsigned short*)alloc((size_t)rowsX * MID * 2);
    unsigned short* kvb   = (unsigned short*)alloc((size_t)rowsX * 2 * MID * 2);
    unsigned short* kvc   = (unsigned short*)alloc((size_t)rowsC * 2 * MID * 2);
    unsigned short* ao    = (unsigned short*)alloc((size_t)rowsX * MID * 2);
    float*          x1    = (float*)alloc((size_t)rowsX * F * 4);
    float*          tabS  = (float*)alloc((size_t)(N + N - 1) * H * 4);
    float*          tabC  = (float*)alloc((size_t)(N + M - 1) * H * 4);

    auto cvt = [&](const float* src, unsigned short* dst, int nelem) {
        cvt_f32_bf16<<<(nelem + 255) / 256, 256, 0, stream>>>(src, dst, nelem);
    };
    cvt(sa_wq,  wq_s,  F * MID);
    cvt(sa_wkv, wkv_s, F * 2 * MID);
    cvt(sa_wo,  wo_s,  MID * F);
    cvt(ca_wq,  wq_c,  F * MID);
    cvt(ca_wkv, wkv_c, F * 2 * MID);
    cvt(ca_wo,  wo_c,  MID * F);

    relbias_kernel<<<(((N + N - 1) * H) + 255) / 256, 256, 0, stream>>>(sa_rel, tabS, N, N);
    relbias_kernel<<<(((N + M - 1) * H) + 255) / 256, 256, 0, stream>>>(ca_rel, tabC, N, M);

    // ---------------- self-attention ----------------
    ln_bf16_kernel<<<rowsX, 256, 0, stream>>>(x, sa_ng,  sa_nb,  xn);
    ln_bf16_kernel<<<rowsX, 256, 0, stream>>>(x, sa_ncg, sa_ncb, cn);

    gemm_bf16_kernel<<<dim3(MID / BN, rowsX / BM), 256, 0, stream>>>(
        xn, wq_s, rowsX, MID, F, qb, nullptr, nullptr, nullptr, 0);
    gemm_bf16_kernel<<<dim3(2 * MID / BN, rowsX / BM), 256, 0, stream>>>(
        cn, wkv_s, rowsX, 2 * MID, F, kvb, nullptr, nullptr, nullptr, 0);

    attn_kernel<<<dim3(N / 64, H, B), 128, 0, stream>>>(qb, kvb, tabS, ao, N, N);

    gemm_bf16_kernel<<<dim3(F / BN, rowsX / BM), 256, 0, stream>>>(
        ao, wo_s, rowsX, F, MID, nullptr, x1, sa_bo, x, 1);

    // ---------------- cross-attention ----------------
    ln_bf16_kernel<<<rowsX, 256, 0, stream>>>(x1, ca_ng, ca_nb, xn);
    ln_bf16_kernel<<<rowsC, 256, 0, stream>>>(context, ca_ncg, ca_ncb, cnx);

    gemm_bf16_kernel<<<dim3(MID / BN, rowsX / BM), 256, 0, stream>>>(
        xn, wq_c, rowsX, MID, F, qb, nullptr, nullptr, nullptr, 0);
    gemm_bf16_kernel<<<dim3(2 * MID / BN, rowsC / BM), 256, 0, stream>>>(
        cnx, wkv_c, rowsC, 2 * MID, F, kvc, nullptr, nullptr, nullptr, 0);

    attn_kernel<<<dim3(N / 64, H, B), 128, 0, stream>>>(qb, kvc, tabC, ao, N, M);

    gemm_bf16_kernel<<<dim3(F / BN, rowsX / BM), 256, 0, stream>>>(
        ao, wo_c, rowsX, F, MID, nullptr, out, ca_bo, x1, 1);
}
